// RuleListNet_87531433492873
// MI455X (gfx1250) — compile-verified, matrix operand order checked by reading, not run on hardware
//
#include <hip/hip_runtime.h>
#include <hip/hip_bf16.h>

typedef __attribute__((ext_vector_type(16))) _Float16 v16h;
typedef __attribute__((ext_vector_type(8)))  float    v8f;

#define B_ROWS   32768
#define D_DIM    256
#define K_THR    6
#define R_RULES  512          // N dimension of big GEMM
#define O_OUT    8
#define KTOT     3072         // 2*D*K  (GEMM K dimension)
#define EPS_F    1e-6f
#define INV_TAU  (1.0f / 0.7f)
#define K_STEPS  96           // 3072 / 32
#define CHUNKS   12           // 96 k-steps / 8 per chunk
#define KS_PER_CHUNK 8
#define M_ROWS   32           // rows per block (2 M-tiles)

// WMMA 16-bit fragment k-map: lane L, vector element h -> k within a 32-wide step.
__device__ __forceinline__ int kmap(int lane, int h) {
    int j = h >> 1, p = h & 1;
    return ((j >> 2) << 4) + ((lane >> 4) << 3) + ((j & 3) << 1) + p;
}

// ---------------------------------------------------------------------------
// Kernel 1: pack m = sigmoid(W / tau) as f16 in WMMA B-fragment order.
// bp layout: [n_tile][k_step][lane][16 halves]  (one v16h per lane per frag)
// ---------------------------------------------------------------------------
__global__ __launch_bounds__(256) void bpack_kernel(const float* __restrict__ W,
                                                    _Float16* __restrict__ bp) {
    int idx  = blockIdx.x * 256 + threadIdx.x;      // one thread per half
    int h    = idx & 15;
    int lane = (idx >> 4) & 31;
    int rem  = idx >> 9;                            // 0..3071
    int ks   = rem % K_STEPS;
    int nt   = rem / K_STEPS;                       // 0..31
    int k    = ks * 32 + kmap(lane, h);
    int n    = nt * 16 + (lane & 15);
    float w  = W[n * KTOT + k];
    float mv = 1.0f / (1.0f + __expf(-w * INV_TAU));
    bp[idx]  = (_Float16)mv;
}

// ---------------------------------------------------------------------------
// Kernel 2: fused feature-gen + WMMA GEMM + exp/clip + cumprod scan + epilogue.
// One block = 256 threads = 8 waves; handles 32 rows of x, all 512 rules.
// Each wave: 2 M-tiles x 4 N-tiles -> one B fragment feeds 2 WMMAs.
// ---------------------------------------------------------------------------
__global__ __launch_bounds__(256) void fused_kernel(const float* __restrict__ x,
                                                    const float* __restrict__ thr,
                                                    const float* __restrict__ log_beta,
                                                    const float* __restrict__ v,
                                                    const float* __restrict__ v_def,
                                                    const _Float16* __restrict__ bp,
                                                    float* __restrict__ out) {
    __shared__ float x_s[M_ROWS * D_DIM];        // 32 KB
    __shared__ float th_s[D_DIM * K_THR];        // 6 KB
    __shared__ float be_s[D_DIM * K_THR];        // 6 KB
    __shared__ float v_s[R_RULES * O_OUT];       // 16 KB
    __shared__ v16h  A_s[KS_PER_CHUNK][2][32];   // 16 KB, A frags (2 M-tiles)
    __shared__ float z_s[R_RULES * M_ROWS];      // 64 KB, z[n][m]

    const int t    = threadIdx.x;
    const int lane = t & 31;
    const int wave = t >> 5;
    const int row0 = blockIdx.x * M_ROWS;

    // ---- stage x tile (32x256 f32) and v (512x8 f32) into LDS (float4 loads)
    {
        const float4* xg = (const float4*)(x + (size_t)row0 * D_DIM);
        float4*       xs = (float4*)x_s;
        const float4* vg = (const float4*)v;
        float4*       vs = (float4*)v_s;
#pragma unroll
        for (int i = 0; i < 8; ++i)
            xs[t + i * 256] = xg[t + i * 256];   // 2048 float4 total
#pragma unroll
        for (int i = 0; i < 4; ++i)
            vs[t + i * 256] = vg[t + i * 256];   // 1024 float4 total
        for (int i = t; i < D_DIM * K_THR; i += 256) {
            th_s[i]  = thr[i];
            float lb = log_beta[i];
            be_s[i]  = (lb > 15.0f) ? lb : __logf(1.0f + __expf(lb));  // softplus
        }
    }
    __syncthreads();

    const v16h* Bp = (const v16h*)bp;
    // hint the B stream for this wave's first n-tile into cache
    __builtin_prefetch(bp + ((size_t)(wave * 4) * K_STEPS * 32) * 16, 0, 0);

    v8f zero8 = {0.f, 0.f, 0.f, 0.f, 0.f, 0.f, 0.f, 0.f};
    v8f acc[2][4];
#pragma unroll
    for (int mt = 0; mt < 2; ++mt)
#pragma unroll
        for (int nt = 0; nt < 4; ++nt) acc[mt][nt] = zero8;

    for (int c = 0; c < CHUNKS; ++c) {
        // ---- produce A fragments for this 256-wide K chunk.
        // thread (wave=ks, lane) fills 16 halves for each of the 2 M-tiles.
#pragma unroll
        for (int mt = 0; mt < 2; ++mt) {
            const int m = mt * 16 + (lane & 15);
            _Float16* ap = (_Float16*)&A_s[wave][mt][lane];
#pragma unroll
            for (int h = 0; h < 16; ++h) {
                int kg     = c * 256 + wave * 32 + kmap(lane, h);
                int second = kg >= (KTOT / 2);
                int pos    = second ? kg - (KTOT / 2) : kg;    // d*K + kk
                float xv   = x_s[m * D_DIM + pos / K_THR];
                float f    = 1.0f / (1.0f + __expf(-(xv - th_s[pos]) * be_s[pos]));
                float fa   = second ? (1.0f - f) : f;
                ap[h]      = (_Float16)__logf(fa + EPS_F);
            }
        }
        __syncthreads();

        // ---- WMMA: 8 k-steps; per k-step: 4 B fragments, each feeds 2 WMMAs
#pragma unroll 1
        for (int ks = 0; ks < KS_PER_CHUNK; ++ks) {
            v16h a0 = A_s[ks][0][lane];
            v16h a1 = A_s[ks][1][lane];
            int kglob = c * KS_PER_CHUNK + ks;
#pragma unroll
            for (int nt = 0; nt < 4; ++nt) {
                int ntg = wave * 4 + nt;
                v16h b = Bp[((size_t)ntg * K_STEPS + kglob) * 32 + lane];
                acc[0][nt] = __builtin_amdgcn_wmma_f32_16x16x32_f16(
                    false, a0, false, b, (short)0, acc[0][nt], false, false);
                acc[1][nt] = __builtin_amdgcn_wmma_f32_16x16x32_f16(
                    false, a1, false, b, (short)0, acc[1][nt], false, false);
            }
        }
        __syncthreads();   // before next chunk overwrites A_s
    }

    // ---- epilogue: z = clip(exp(acc), 0, 1) -> z_s[n][m]
    {
        int mbase = (lane >> 4) * 8;
#pragma unroll
        for (int mt = 0; mt < 2; ++mt) {
#pragma unroll
            for (int nt = 0; nt < 4; ++nt) {
                int n = (wave * 4 + nt) * 16 + (lane & 15);
#pragma unroll
                for (int r = 0; r < 8; ++r) {
                    float zz = __expf(acc[mt][nt][r]);
                    zz = fminf(zz, 1.0f);
                    z_s[n * M_ROWS + mt * 16 + mbase + r] = zz;
                }
            }
        }
    }
    __syncthreads();

    // ---- stop-probability scan + small output GEMM.
    // thread (mm, o): redundant per-o cumprod; all 256 lanes busy, z_s reads
    // broadcast within groups of 8 lanes.
    {
        int mm = t >> 3, o = t & 7;   // mm in [0,32), o in [0,8)
        float p = 1.0f, a0 = 0.0f;
        for (int j = 0; j < R_RULES; ++j) {
            float zz = z_s[j * M_ROWS + mm];
            a0 += zz * p * v_s[j * O_OUT + o];
            p  *= (1.0f - zz + EPS_F);
        }
        out[(size_t)(row0 + mm) * O_OUT + o] = a0 + p * v_def[o];
    }
}

extern "C" void kernel_launch(void* const* d_in, const int* in_sizes, int n_in,
                              void* d_out, int out_size, void* d_ws, size_t ws_size,
                              hipStream_t stream) {
    const float* x   = (const float*)d_in[0];
    const float* thr = (const float*)d_in[1];
    const float* lb  = (const float*)d_in[2];
    const float* W   = (const float*)d_in[3];
    const float* v   = (const float*)d_in[4];
    const float* vd  = (const float*)d_in[5];
    float* out       = (float*)d_out;
    _Float16* bp     = (_Float16*)d_ws;      // 512*3072 halves = 3 MB

    // pack B fragments: 32*96*32*16 = 1,572,864 halves -> 6144 blocks
    bpack_kernel<<<6144, 256, 0, stream>>>(W, bp);
    // fused main: 32768 rows / 32 per block
    fused_kernel<<<B_ROWS / M_ROWS, 256, 0, stream>>>(x, thr, lb, v, vd, bp, out);
}
